// EdgeEncoder_30305289240586
// MI455X (gfx1250) — matrix operand order repaired and checked by reading, not compiled, not run on hardware
//
#include <hip/hip_runtime.h>

typedef __attribute__((ext_vector_type(16))) _Float16 v16h;
typedef __attribute__((ext_vector_type(8)))  _Float16 v8h;
typedef __attribute__((ext_vector_type(8)))  float    v8f;

#define HID 64
#define K1  128
#define LN_EPS 1e-5f

// padded LDS strides (elements) to stagger the 64 LDS banks
#define WT1_STRIDE   136   // f16: 128 + 8 pad
#define WT23_STRIDE  72    // f16: 64 + 8 pad
#define A1_STRIDE    132   // f32: 128 + 4 pad (layer-1 input, async-filled)
#define A2_STRIDE    72    // f16: 64 + 8 pad  (layer-2/3 input)
#define HST_STRIDE   68    // f32: 64 + 4 pad  (C spill for LayerNorm)

__device__ __forceinline__ v8f wmma16x16x32(v16h a, v16h b, v8f c) {
  return __builtin_amdgcn_wmma_f32_16x16x32_f16(
      false, a, false, b, (short)0, c, false, false);
}

// Low 32 bits of a flat LDS address == LDS byte offset (aperture layout).
__device__ __forceinline__ unsigned lds_off32(const void* p) {
  return (unsigned)(unsigned long long)p;
}

// CDNA5 async copy: 16B global -> LDS, tracked on ASYNCcnt (no VGPR round-trip)
__device__ __forceinline__ void async_copy_b128(unsigned lds_byte_off,
                                                const void* gaddr) {
  asm volatile("global_load_async_to_lds_b128 %0, %1, off"
               :: "v"(lds_byte_off), "v"(gaddr) : "memory");
}
__device__ __forceinline__ void wait_async0() {
  asm volatile("s_wait_asynccnt 0x0" ::: "memory");
}

__global__ __launch_bounds__(256) void edge_mlp_wmma(
    const float* __restrict__ x,            // [n_nodes, 64]
    const int*   __restrict__ edge_index,   // [2, n_edges]
    const float* __restrict__ W1, const float* __restrict__ b1,
    const float* __restrict__ g1, const float* __restrict__ be1,
    const float* __restrict__ W2, const float* __restrict__ b2,
    const float* __restrict__ g2, const float* __restrict__ be2,
    const float* __restrict__ W3, const float* __restrict__ b3,
    float* __restrict__ out, int n_edges, int n_tiles)
{
  __shared__ _Float16 s_wt1[HID * WT1_STRIDE];      // W1^T [n][k], f16
  __shared__ _Float16 s_wt2[HID * WT23_STRIDE];     // W2^T
  __shared__ _Float16 s_wt3[HID * WT23_STRIDE];     // W3^T
  __shared__ float    s_bias[7 * HID];              // b1,g1,be1,b2,g2,be2,b3
  __shared__ float    s_a1[8][16 * A1_STRIDE];      // per-wave gather stage (f32)
  __shared__ _Float16 s_a2[8][16 * A2_STRIDE];      // per-wave activ. stage (f16)
  __shared__ float    s_h[8][16 * HST_STRIDE];      // per-wave C spill (f32)

  const int tid = threadIdx.x;

  // ---- one-time (per persistent block) weight/bias staging ----
  for (int i = tid; i < K1 * HID; i += 256) {
    int k = i >> 6, n = i & 63;
    s_wt1[n * WT1_STRIDE + k] = (_Float16)W1[i];
  }
  for (int i = tid; i < HID * HID; i += 256) {
    int k = i >> 6, n = i & 63;
    s_wt2[n * WT23_STRIDE + k] = (_Float16)W2[i];
    s_wt3[n * WT23_STRIDE + k] = (_Float16)W3[i];
  }
  if (tid < HID) {
    s_bias[          tid] = b1[tid];
    s_bias[1 * HID + tid] = g1[tid];
    s_bias[2 * HID + tid] = be1[tid];
    s_bias[3 * HID + tid] = b2[tid];
    s_bias[4 * HID + tid] = g2[tid];
    s_bias[5 * HID + tid] = be2[tid];
    s_bias[6 * HID + tid] = b3[tid];
  }
  __syncthreads();

  const int wave  = tid >> 5;
  const int lane  = tid & 31;
  const int row16 = lane & 15;
  const int hk    = lane >> 4;

  float*    A1 = s_a1[wave];
  _Float16* A2 = s_a2[wave];
  float*    H  = s_h[wave];

  // ---- fragment loaders (documented 16-bit WMMA VGPR layouts) ----
  // A 16x32, from f32 stage: lanes<16: elems0-7=K+0..7, 8-15=K+16..23; lanes>=16: +8
  auto load_a1 = [&](int kb) -> v16h {
    const float* base = A1 + row16 * A1_STRIDE + kb + hk * 8;
    float4 p0 = *(const float4*)(base);
    float4 p1 = *(const float4*)(base + 4);
    float4 q0 = *(const float4*)(base + 16);
    float4 q1 = *(const float4*)(base + 20);
    v16h a;
    a[0]=(_Float16)p0.x;  a[1]=(_Float16)p0.y;  a[2]=(_Float16)p0.z;  a[3]=(_Float16)p0.w;
    a[4]=(_Float16)p1.x;  a[5]=(_Float16)p1.y;  a[6]=(_Float16)p1.z;  a[7]=(_Float16)p1.w;
    a[8]=(_Float16)q0.x;  a[9]=(_Float16)q0.y;  a[10]=(_Float16)q0.z; a[11]=(_Float16)q0.w;
    a[12]=(_Float16)q1.x; a[13]=(_Float16)q1.y; a[14]=(_Float16)q1.z; a[15]=(_Float16)q1.w;
    return a;
  };
  auto load_a2 = [&](int kb) -> v16h {
    const _Float16* base = A2 + row16 * A2_STRIDE + kb + hk * 8;
    v8h lo = *(const v8h*)(base);
    v8h hi = *(const v8h*)(base + 16);
    v16h a;
    #pragma unroll
    for (int i = 0; i < 8; ++i) { a[i] = lo[i]; a[8 + i] = hi[i]; }
    return a;
  };
  // B 32x16: lane owns column N=nb+row16; elems0-7 = K kb+hk*8..+7, 8-15 = K kb+16+hk*8..+7
  auto load_b = [&](const _Float16* WT, int stride, int nb, int kb) -> v16h {
    const _Float16* p = WT + (nb + row16) * stride + kb + hk * 8;
    v8h lo = *(const v8h*)(p);
    v8h hi = *(const v8h*)(p + 16);
    v16h b;
    #pragma unroll
    for (int i = 0; i < 8; ++i) { b[i] = lo[i]; b[8 + i] = hi[i]; }
    return b;
  };
  auto store_c = [&](v8f acc, int nb) {
    #pragma unroll
    for (int r = 0; r < 8; ++r)
      H[(r + hk * 8) * HST_STRIDE + nb + row16] = acc[r];
  };
  // fused bias + LayerNorm + SiLU; repack result as f16 into A2
  auto ln_silu = [&](int boff) {
    const float* bia = s_bias + boff;
    const float* gam = s_bias + boff + HID;
    const float* bet = s_bias + boff + 2 * HID;
    const float* hp  = H + row16 * HST_STRIDE + hk * 32;
    float v[32], s = 0.f, s2 = 0.f;
    #pragma unroll
    for (int i = 0; i < 32; ++i) {
      float t = hp[i] + bia[hk * 32 + i];
      v[i] = t; s += t; s2 += t * t;
    }
    s  += __shfl_xor(s, 16, 32);
    s2 += __shfl_xor(s2, 16, 32);
    const float mu  = s * (1.f / 64.f);
    const float var = s2 * (1.f / 64.f) - mu * mu;
    const float rs  = rsqrtf(var + LN_EPS);
    _Float16* ap = A2 + row16 * A2_STRIDE + hk * 32;
    #pragma unroll
    for (int i = 0; i < 32; ++i) {
      float t = (v[i] - mu) * rs * gam[hk * 32 + i] + bet[hk * 32 + i];
      float sg = 1.f / (1.f + __expf(-t));
      ap[i] = (_Float16)(t * sg);
    }
  };

  // ---- persistent-block loop over 16-edge tiles (8 tiles per block pass) ----
  for (int tile = blockIdx.x; tile < n_tiles; tile += gridDim.x) {
    const long long ebase = (long long)tile * 128 + (long long)wave * 16;

    // warm next pass's edge indices into cache (CDNA5 global_prefetch)
    {
      long long pe = ebase + (long long)gridDim.x * 128;
      if (pe < n_edges && lane == 0) {
        __builtin_prefetch(edge_index + pe, 0, 0);
        __builtin_prefetch(edge_index + n_edges + pe, 0, 0);
      }
    }

    // ---- async gather: x[src]||x[dst] -> f32 A1 stage [16][128] ----
    // lanes 0..15 copy src cols, lanes 16..31 copy dst cols, 16B per lane/row
    #pragma unroll 4
    for (int r = 0; r < 16; ++r) {
      long long e = ebase + r;
      if (e >= n_edges) e = n_edges - 1;            // clamp (uniform per wave)
      const int node = (hk == 0) ? edge_index[e]
                                 : edge_index[(long long)n_edges + e];
      const float* src = x + (long long)node * HID + row16 * 4;
      unsigned dst = lds_off32(A1 + r * A1_STRIDE + hk * 64 + row16 * 4);
      async_copy_b128(dst, src);
    }
    wait_async0();                                  // drain ASYNCcnt

    // ---- layer 1: [16x128] @ W1 -> H ----
    {
      v16h a0 = load_a1(0), a1 = load_a1(32), a2 = load_a1(64), a3 = load_a1(96);
      #pragma unroll
      for (int nt = 0; nt < 4; ++nt) {
        v8f acc = {};
        acc = wmma16x16x32(a0, load_b(s_wt1, WT1_STRIDE, nt * 16,  0), acc);
        acc = wmma16x16x32(a1, load_b(s_wt1, WT1_STRIDE, nt * 16, 32), acc);
        acc = wmma16x16x32(a2, load_b(s_wt1, WT1_STRIDE, nt * 16, 64), acc);
        acc = wmma16x16x32(a3, load_b(s_wt1, WT1_STRIDE, nt * 16, 96), acc);
        store_c(acc, nt * 16);
      }
    }
    ln_silu(0);

    // ---- layer 2: [16x64] @ W2 -> H ----
    {
      v16h a0 = load_a2(0), a1 = load_a2(32);
      #pragma unroll
      for (int nt = 0; nt < 4; ++nt) {
        v8f acc = {};
        acc = wmma16x16x32(a0, load_b(s_wt2, WT23_STRIDE, nt * 16,  0), acc);
        acc = wmma16x16x32(a1, load_b(s_wt2, WT23_STRIDE, nt * 16, 32), acc);
        store_c(acc, nt * 16);
      }
    }
    ln_silu(3 * HID);

    // ---- layer 3: [16x64] @ W3 + b3 -> global ----
    {
      v16h a0 = load_a2(0), a1 = load_a2(32);
      #pragma unroll
      for (int nt = 0; nt < 4; ++nt) {
        v8f acc = {};
        acc = wmma16x16x32(a0, load_b(s_wt3, WT23_STRIDE, nt * 16,  0), acc);
        acc = wmma16x16x32(a1, load_b(s_wt3, WT23_STRIDE, nt * 16, 32), acc);
        const int n  = nt * 16 + row16;
        const float bb = s_bias[6 * HID + n];
        #pragma unroll
        for (int r = 0; r < 8; ++r) {
          long long e = ebase + r + hk * 8;
          if (e < n_edges) out[e * HID + n] = acc[r] + bb;
        }
      }
    }
  }
}

extern "C" void kernel_launch(void* const* d_in, const int* in_sizes, int n_in,
                              void* d_out, int out_size, void* d_ws, size_t ws_size,
                              hipStream_t stream) {
  const float* x          = (const float*)d_in[0];
  const int*   edge_index = (const int*)d_in[1];   // int32 per harness convention
  const float* W1  = (const float*)d_in[2];
  const float* b1  = (const float*)d_in[3];
  const float* g1  = (const float*)d_in[4];
  const float* be1 = (const float*)d_in[5];
  const float* W2  = (const float*)d_in[6];
  const float* b2  = (const float*)d_in[7];
  const float* g2  = (const float*)d_in[8];
  const float* be2 = (const float*)d_in[9];
  const float* W3  = (const float*)d_in[10];
  const float* b3  = (const float*)d_in[11];
  float* out = (float*)d_out;

  const int n_edges = in_sizes[1] / 2;
  const int n_tiles = (n_edges + 127) / 128;        // 128 edges per 256-thr block
  int blocks = n_tiles < 4096 ? n_tiles : 4096;     // persistent blocks

  edge_mlp_wmma<<<blocks, 256, 0, stream>>>(
      x, edge_index, W1, b1, g1, be1, W2, b2, g2, be2, W3, b3,
      out, n_edges, n_tiles);
}
